// Head_71004399337653
// MI455X (gfx1250) — matrix operand order repaired and checked by reading, not compiled, not run on hardware
//
#include <hip/hip_runtime.h>
#include <hip/hip_bf16.h>

typedef __attribute__((ext_vector_type(16))) __bf16 v16bf;
typedef __attribute__((ext_vector_type(8)))  float  v8f;

#define BATCH 4
#define SEQ   4096
#define CDIM  256
#define HDIM  64
#define QSCALE 0.0625f   /* CDIM^-0.5 = 1/16 exactly */
#define LPAD  72         /* LDS row stride (64 + 8 pad) -> conflict-free b128 reads */

union BF16x16 { v16bf v; uint4 q[2]; };
union BF16x8  { uint4 q; __bf16 e[8]; };

// D = A(16x32 bf16) x B(32x16 bf16) + C(16x16 f32)
__device__ __forceinline__ v8f wmma_bf16(v16bf a, v16bf b, v8f c) {
  return __builtin_amdgcn_wmma_f32_16x16x32_bf16(false, a, false, b,
                                                 (short)0, c, false, false);
}

// ---------------------------------------------------------------------------
// Tensor Data Mover path (gfx1250): one descriptor loads a 64x64 bf16 tile,
// padding each 128B row with 16B so LDS rows land on the LPAD=72 layout.
// This toolchain exposes the 6-arg builtin:
//   (uint32x4 g0, int32x8 g1, int32x4 g2, int32x4 g3, int32x8 g4, i32 cpol)
// ---------------------------------------------------------------------------
#if defined(__has_builtin)
#if __has_builtin(__builtin_amdgcn_tensor_load_to_lds) && \
    __has_builtin(__builtin_amdgcn_s_wait_tensorcnt)
#define HAVE_TDM 1
#endif
#endif

#if defined(HAVE_TDM)
typedef __attribute__((ext_vector_type(4))) unsigned int u32x4;
typedef __attribute__((ext_vector_type(8))) int          i32x8;
typedef __attribute__((ext_vector_type(4))) int          i32x4;

__device__ __forceinline__ void tdm_load_tile_64x64(unsigned lds_off,
                                                    unsigned long long gaddr,
                                                    unsigned stride_elems) {
  u32x4 g0;
  g0[0] = 1u;                                              // count=1, no gather
  g0[1] = lds_off;                                         // lds_addr[63:32]
  g0[2] = (unsigned)(gaddr & 0xFFFFFFFFu);                 // global_addr lo
  g0[3] = (unsigned)((gaddr >> 32) & 0x1FFFFFFu)           // global_addr hi
        | (2u << 30);                                      // type=2 ("image")
  i32x8 g1;
  g1[0] = (int)((1u << 16)      /* data_size = 2B */
              | (1u << 20)      /* pad_enable */
              | (4u << 22)      /* pad_interval: 32 DWORDs (one 128B row) */
              | (3u << 25));    /* pad_amount: 4 DWORDs (16B -> LPAD) */
  g1[1] = (int)(64u << 16);     // tensor_dim0 = 64 (atomic_barrier_addr = 0)
  g1[2] = (int)(64u << 16);     // tensor_dim1 = 64
  g1[3] = (int)(64u << 16);     // tile_dim0 = 64
  g1[4] = 64;                   // tile_dim1 = 64, tile_dim2 = 0
  g1[5] = (int)stride_elems;    // tensor_dim0_stride (lo32)
  g1[6] = 0;
  g1[7] = 0;
  i32x4 z4 = {0, 0, 0, 0};                   // groups 2/3 unused for 2D tiles
  i32x8 z8 = {0, 0, 0, 0, 0, 0, 0, 0};       // extra group unused
  __builtin_amdgcn_tensor_load_to_lds(g0, g1, z4, z4, z8, 0);
}
#endif

// ---------------------------------------------------------------------------
// One-time prep: x (f32) -> bf16, W (f32 [C][H]) -> bf16 W^T [H][C]
// ---------------------------------------------------------------------------
__global__ __launch_bounds__(256) void cast_x(const float* __restrict__ x,
                                              __bf16* __restrict__ xb) {
  const size_t i = ((size_t)blockIdx.x * 256 + threadIdx.x) * 8;
  float4 f0 = *(const float4*)(x + i);
  float4 f1 = *(const float4*)(x + i + 4);
  BF16x8 u;
  u.e[0] = (__bf16)f0.x; u.e[1] = (__bf16)f0.y;
  u.e[2] = (__bf16)f0.z; u.e[3] = (__bf16)f0.w;
  u.e[4] = (__bf16)f1.x; u.e[5] = (__bf16)f1.y;
  u.e[6] = (__bf16)f1.z; u.e[7] = (__bf16)f1.w;
  *(uint4*)(xb + i) = u.q;
}

__global__ __launch_bounds__(256) void prep_w(const float* __restrict__ W,
                                              __bf16* __restrict__ Wt) {
  const int t  = threadIdx.x;
  const int h  = t >> 2;
  const int c0 = (t & 3) * 64;
  for (int c = 0; c < 64; ++c)
    Wt[(size_t)h * CDIM + c0 + c] = (__bf16)W[(size_t)(c0 + c) * HDIM + h];
}

// ---------------------------------------------------------------------------
// Projection: (xb[BT x 256] @ Wt^T) * oscale -> bf16.
// TRANSPOSED=false: out[row][h] (row-major, for q and k).
// TRANSPOSED=true : out[b][h][t] (for v: makes the flash V fill a straight
//                   2D tile copy and the store is a single b128 per N-tile).
// ---------------------------------------------------------------------------
template <bool TRANSPOSED>
__global__ __launch_bounds__(32) void proj16(const __bf16* __restrict__ xb,
                                             const __bf16* __restrict__ Wt,
                                             __bf16* __restrict__ out,
                                             float oscale) {
  const int lane = threadIdx.x & 31;
  const int n16  = lane & 15;
  const int hi   = lane >> 4;
  const int row0 = blockIdx.x * 16;

  v8f acc[4] = {};

  for (int k0 = 0; k0 < CDIM; k0 += 32) {
    BF16x16 a;  // A(m = n16, K): comps 0-7 -> K = 8*hi+0..7, 8-15 -> +16
    const __bf16* xr = xb + (size_t)(row0 + n16) * CDIM + k0;
    a.q[0] = *(const uint4*)(xr + 8 * hi);
    a.q[1] = *(const uint4*)(xr + 16 + 8 * hi);
#pragma unroll
    for (int nt = 0; nt < 4; ++nt) {
      BF16x16 b;  // B(K, n = n16): comps 0-7 -> K = 8*hi+0..7, 8-15 -> +16 (K-major Wt)
      const __bf16* wr = Wt + (size_t)(nt * 16 + n16) * CDIM + k0 + 8 * hi;
      b.q[0] = *(const uint4*)(wr);
      b.q[1] = *(const uint4*)(wr + 16);
      acc[nt] = wmma_bf16(a.v, b.v, acc[nt]);
    }
  }

  if (!TRANSPOSED) {
#pragma unroll
    for (int nt = 0; nt < 4; ++nt)
#pragma unroll
      for (int cc = 0; cc < 8; ++cc)
        out[(size_t)(row0 + cc + 8 * hi) * HDIM + nt * 16 + n16] =
            (__bf16)(acc[nt][cc] * oscale);
  } else {
    // components cc are consecutive t values for fixed h -> one b128 per nt
    const int bb2 = row0 / SEQ;
    const int t0  = (row0 % SEQ) + 8 * hi;
#pragma unroll
    for (int nt = 0; nt < 4; ++nt) {
      BF16x8 u;
#pragma unroll
      for (int cc = 0; cc < 8; ++cc) u.e[cc] = (__bf16)(acc[nt][cc] * oscale);
      *(uint4*)(out + ((size_t)bb2 * HDIM + nt * 16 + n16) * SEQ + t0) = u.q;
    }
  }
}

// ---------------------------------------------------------------------------
// Flash attention: 4 waves/block, 32 query rows/wave (128/block).
// K[kv][h] and V^T[h][t] are both straight 2D tile copies into LDS
// (TDM descriptors when available), rows padded to LPAD elements.
// ---------------------------------------------------------------------------
__global__ __launch_bounds__(128) void flash_attn(const __bf16* __restrict__ q,
                                                  const __bf16* __restrict__ k,
                                                  const __bf16* __restrict__ vt,
                                                  float* __restrict__ out) {
  __shared__ __align__(16) __bf16 ldsK[64 * LPAD];      // K[kv][h]
  __shared__ __align__(16) __bf16 ldsVt[64 * LPAD];     // V^T[h][kv]
  __shared__ __align__(16) __bf16 ldsP[4][32 * LPAD];   // per-wave P (32 rows)

  const int tid   = threadIdx.x;
  const int wave  = tid >> 5;
  const int lane  = tid & 31;
  const int n16   = lane & 15;
  const int hi    = lane >> 4;
  const int bb    = blockIdx.y;
  const int qrow0 = blockIdx.x * 128 + wave * 32;

  // Q tile (32 x 64) resident as 2 msub x 2 kc A-operands, 2 x b128 each.
  v16bf aQ[2][2];
#pragma unroll
  for (int ms = 0; ms < 2; ++ms) {
    const __bf16* qr = q + ((size_t)bb * SEQ + qrow0 + ms * 16 + n16) * HDIM;
#pragma unroll
    for (int kc = 0; kc < 2; ++kc) {
      BF16x16 a;
      a.q[0] = *(const uint4*)(qr + kc * 32 + 8 * hi);
      a.q[1] = *(const uint4*)(qr + kc * 32 + 16 + 8 * hi);
      aQ[ms][kc] = a.v;
    }
  }

  v8f o[2][4] = {};
  float mrow[2][8], lrow[2][8];
#pragma unroll
  for (int ms = 0; ms < 2; ++ms)
#pragma unroll
    for (int i = 0; i < 8; ++i) { mrow[ms][i] = -__builtin_inff(); lrow[ms][i] = 0.f; }

  const int jmax = 2 * (int)blockIdx.x + 1;
  for (int j = 0; j <= jmax; ++j) {
    const int kv0 = j * 64;
    const __bf16* kbase = k  + ((size_t)bb * SEQ + kv0) * HDIM;
    const __bf16* vbase = vt + (size_t)bb * HDIM * SEQ + kv0;

    __syncthreads();  // previous iteration done with LDS tiles

    if (j < jmax) {   // prefetch next KV tile into cache (global_prefetch_b8)
      __builtin_prefetch(kbase + (size_t)64 * HDIM + (tid >> 1) * HDIM + (tid & 1) * 32, 0, 3);
      __builtin_prefetch(vbase + 64 + (size_t)(tid >> 1) * SEQ + (tid & 1) * 32, 0, 3);
    }

#if defined(HAVE_TDM)
    if (wave == 0) {
      tdm_load_tile_64x64((unsigned)(unsigned long long)(uintptr_t)&ldsK[0],
                          (unsigned long long)(uintptr_t)kbase, HDIM);
      tdm_load_tile_64x64((unsigned)(unsigned long long)(uintptr_t)&ldsVt[0],
                          (unsigned long long)(uintptr_t)vbase, SEQ);
      __builtin_amdgcn_s_wait_tensorcnt(0);
    }
#else
    {
      const int r    = tid >> 1;        // tile row 0..63
      const int half = (tid & 1) * 32;  // column offset 0 / 32
      const uint4* kg = (const uint4*)(kbase + (size_t)r * HDIM + half);
      const uint4* vg = (const uint4*)(vbase + (size_t)r * SEQ + half);
      uint4* kd = (uint4*)(&ldsK[r * LPAD + half]);
      uint4* vd = (uint4*)(&ldsVt[r * LPAD + half]);
#pragma unroll
      for (int i = 0; i < 4; ++i) { kd[i] = kg[i]; vd[i] = vg[i]; }
    }
#endif
    __syncthreads();

    const bool active = (kv0 <= qrow0 + 31);  // wave-uniform causal tile skip

    if (active) {
      // ---- S = Q K^T : two 16x64 tiles per wave, shared bK loads ----
      v8f sv[2][4];
#pragma unroll
      for (int nt = 0; nt < 4; ++nt) {
        v8f s0 = {}, s1 = {};
#pragma unroll
        for (int kc = 0; kc < 2; ++kc) {
          BF16x16 bK;  // B(k=h, n=kv): two 8-elem h runs in ldsK row
          const __bf16* kr = &ldsK[(nt * 16 + n16) * LPAD + kc * 32 + 8 * hi];
          bK.q[0] = *(const uint4*)(kr);
          bK.q[1] = *(const uint4*)(kr + 16);
          s0 = wmma_bf16(aQ[0][kc], bK.v, s0);
          s1 = wmma_bf16(aQ[1][kc], bK.v, s1);
        }
        sv[0][nt] = s0;
        sv[1][nt] = s1;
      }

      // ---- causal mask (only tiles that straddle the diagonal) ----
#pragma unroll
      for (int ms = 0; ms < 2; ++ms) {
        if (kv0 + 63 > qrow0 + ms * 16) {
#pragma unroll
          for (int nt = 0; nt < 4; ++nt)
#pragma unroll
            for (int cc = 0; cc < 8; ++cc)
              if (kv0 + nt * 16 + n16 > qrow0 + ms * 16 + cc + 8 * hi)
                sv[ms][nt][cc] = -__builtin_inff();
        }
      }

      // ---- online softmax; row = (ms, cc, hi), 16 cols across lanes ----
#pragma unroll
      for (int ms = 0; ms < 2; ++ms) {
#pragma unroll
        for (int cc = 0; cc < 8; ++cc) {
          float smax = sv[ms][0][cc];
#pragma unroll
          for (int nt = 1; nt < 4; ++nt) smax = fmaxf(smax, sv[ms][nt][cc]);
#pragma unroll
          for (int off = 1; off < 16; off <<= 1)
            smax = fmaxf(smax, __shfl_xor(smax, off, 32));

          const float mnew = fmaxf(mrow[ms][cc], smax);
          const float corr = __expf(mrow[ms][cc] - mnew);
          float rs = 0.f;
#pragma unroll
          for (int nt = 0; nt < 4; ++nt) {
            float p = __expf(sv[ms][nt][cc] - mnew);
            sv[ms][nt][cc] = p;
            rs += p;
          }
#pragma unroll
          for (int off = 1; off < 16; off <<= 1)
            rs += __shfl_xor(rs, off, 32);

          lrow[ms][cc] = lrow[ms][cc] * corr + rs;
          mrow[ms][cc] = mnew;
#pragma unroll
          for (int nt = 0; nt < 4; ++nt) o[ms][nt][cc] *= corr;
        }
      }

      // ---- P: D-layout -> A-layout via per-wave LDS buffer (32 rows) ----
      __bf16* P = ldsP[wave];
#pragma unroll
      for (int ms = 0; ms < 2; ++ms)
#pragma unroll
        for (int nt = 0; nt < 4; ++nt)
#pragma unroll
          for (int cc = 0; cc < 8; ++cc)
            P[(ms * 16 + cc + 8 * hi) * LPAD + nt * 16 + n16] = (__bf16)sv[ms][nt][cc];
    }
    __syncthreads();

    if (active) {
      // ---- O += P(32x64) V(64x64), shared bV loads ----
      const __bf16* P = ldsP[wave];
#pragma unroll
      for (int kc = 0; kc < 2; ++kc) {
        BF16x16 aP0, aP1;  // A(m=n16, K=kv): two contiguous 8-elem runs
        const __bf16* pr0 = &P[n16 * LPAD + kc * 32 + 8 * hi];
        const __bf16* pr1 = &P[(16 + n16) * LPAD + kc * 32 + 8 * hi];
        aP0.q[0] = *(const uint4*)(pr0);
        aP0.q[1] = *(const uint4*)(pr0 + 16);
        aP1.q[0] = *(const uint4*)(pr1);
        aP1.q[1] = *(const uint4*)(pr1 + 16);
#pragma unroll
        for (int nt = 0; nt < 4; ++nt) {
          BF16x16 bV;  // B(k=kv, n=h): two 8-elem kv runs in ldsVt row
          const __bf16* vr = &ldsVt[(nt * 16 + n16) * LPAD + kc * 32 + 8 * hi];
          bV.q[0] = *(const uint4*)(vr);
          bV.q[1] = *(const uint4*)(vr + 16);
          o[0][nt] = wmma_bf16(aP0.v, bV.v, o[0][nt]);
          o[1][nt] = wmma_bf16(aP1.v, bV.v, o[1][nt]);
        }
      }
    }
  }

  // ---- epilogue: normalize by running sum, store f32 ----
#pragma unroll
  for (int ms = 0; ms < 2; ++ms)
#pragma unroll
    for (int nt = 0; nt < 4; ++nt)
#pragma unroll
      for (int cc = 0; cc < 8; ++cc)
        out[((size_t)bb * SEQ + qrow0 + ms * 16 + cc + 8 * hi) * HDIM + nt * 16 + n16] =
            o[ms][nt][cc] / lrow[ms][cc];
}

// ---------------------------------------------------------------------------
extern "C" void kernel_launch(void* const* d_in, const int* in_sizes, int n_in,
                              void* d_out, int out_size, void* d_ws, size_t ws_size,
                              hipStream_t stream) {
  (void)in_sizes; (void)n_in; (void)out_size; (void)ws_size;

  const float* x  = (const float*)d_in[0];
  const float* Wk = (const float*)d_in[1];
  const float* Wq = (const float*)d_in[2];
  const float* Wv = (const float*)d_in[3];
  float* out = (float*)d_out;

  const size_t BT = (size_t)BATCH * SEQ;
  __bf16* qb  = (__bf16*)d_ws;            // [BT x 64]     2 MB
  __bf16* kb  = qb  + BT * HDIM;          // [BT x 64]     2 MB
  __bf16* vtb = kb  + BT * HDIM;          // [B][64][SEQ]  2 MB (V transposed)
  __bf16* xb  = vtb + BT * HDIM;          // [BT x 256]    8 MB
  __bf16* wtq = xb  + BT * CDIM;          // [64 x 256]   32 KB
  __bf16* wtk = wtq + (size_t)CDIM * HDIM;
  __bf16* wtv = wtk + (size_t)CDIM * HDIM;

  cast_x<<<dim3((unsigned)(BT * CDIM / (256 * 8))), 256, 0, stream>>>(x, xb);
  prep_w<<<1, 256, 0, stream>>>(Wq, wtq);
  prep_w<<<1, 256, 0, stream>>>(Wk, wtk);
  prep_w<<<1, 256, 0, stream>>>(Wv, wtv);

  dim3 pgrid((unsigned)(BT / 16));
  proj16<false><<<pgrid, 32, 0, stream>>>(xb, wtq, qb, QSCALE);  // 1/sqrt(C) in q
  proj16<false><<<pgrid, 32, 0, stream>>>(xb, wtk, kb, 1.0f);
  proj16<true ><<<pgrid, 32, 0, stream>>>(xb, wtv, vtb, 1.0f);   // V^T layout

  dim3 agrid(SEQ / 128, BATCH);
  flash_attn<<<agrid, 128, 0, stream>>>(qb, kb, vtb, out);
}
